// So3HyperCNNv2_37374805410391
// MI455X (gfx1250) — matrix operand order repaired and checked by reading, not compiled, not run on hardware
//
#include <hip/hip_runtime.h>
#include <hip/hip_bf16.h>

#define EPS 1e-6f
#define LMAX 7
#define NSH 64            // sum(2l+1), l=0..7
#define KTOT 4160         // 64*64 bilinear + 64 linear self-path
#define NCHUNK 130        // KTOT / 32
#define TILE 16

typedef _Float16 v16h __attribute__((ext_vector_type(16)));
typedef float    v8f  __attribute__((ext_vector_type(8)));

// ---------------------------------------------------------------------------
// Workspace layout: folded CG tensor, pre-swizzled into WMMA B-operand order:
//   Tsw[blk][chunk][ntile][lane][elem]  (_Float16)
//   element (j,k):  chunk=j>>5, K=j&31, ntile=k>>4, lane=((K>>4)<<4)|(k&15), e=K&15
// ---------------------------------------------------------------------------
__device__ __host__ inline size_t sw_index(int b, int j, int k) {
  int c = j >> 5, K = j & 31;
  int nt = k >> 4;
  int lane = ((K >> 4) << 4) | (k & 15);
  int e = K & 15;
  return ((((size_t)b * NCHUNK + c) * 4 + nt) * 32 + lane) * 16 + e;
}

// ---- CDNA5 async global->LDS copy (ASYNCcnt path) -------------------------
__device__ inline void async_load_f32_to_lds(float* lds_dst, const float* gsrc) {
  unsigned ldsa = (unsigned)(unsigned long long)(void*)lds_dst;
  unsigned long long ga = (unsigned long long)gsrc;
  asm volatile("global_load_async_to_lds_b32 %0, %1, off"
               :: "v"(ldsa), "v"(ga) : "memory");
}
__device__ inline void wait_async0() {
  asm volatile("s_wait_asynccnt 0x0" ::: "memory");
}

// ------------------------------- fold kernels ------------------------------

__global__ void zero_ws_kernel(unsigned int* p, int n) {
  int i = blockIdx.x * blockDim.x + threadIdx.x;
  if (i < n) p[i] = 0u;
}

__global__ void fold_triple_kernel(unsigned short* Tsw_, const float* C,
                                   const float* wm0, const float* wm1,
                                   const float* wm2,
                                   int l1, int l2, int l3, int pidx) {
  _Float16* Tsw = reinterpret_cast<_Float16*>(Tsw_);
  int n1 = 2 * l1 + 1, n2 = 2 * l2 + 1, n3 = 2 * l3 + 1;
  int per = n1 * n2 * n3;
  int i = blockIdx.x * blockDim.x + threadIdx.x;
  if (i >= 3 * per) return;
  int b = i / per, rem = i % per;
  int m1 = rem / (n2 * n3); rem %= (n2 * n3);
  int m2 = rem / n3, m3 = rem % n3;
  const float* wm = (b == 0) ? wm0 : ((b == 1) ? wm1 : wm2);
  float val = wm[pidx] * C[(m1 * n2 + m2) * n3 + m3];
  int j = (l1 * l1 + m1) * NSH + (l2 * l2 + m2);
  int k = l3 * l3 + m3;
  Tsw[sw_index(b, j, k)] = (_Float16)val;
}

struct WmPtrs { const float* wm[3][8]; };

__global__ void fold_self_kernel(unsigned short* Tsw_, WmPtrs W) {
  _Float16* Tsw = reinterpret_cast<_Float16*>(Tsw_);
  int i = blockIdx.x * blockDim.x + threadIdx.x;
  if (i >= 3 * NSH) return;
  int b = i >> 6, q = i & 63;
  int l = 0;
  while ((l + 1) * (l + 1) <= q) ++l;
  Tsw[sw_index(b, 4096 + q, q)] = (_Float16)W.wm[b][l][0];
}

// ------------------------------- main kernel -------------------------------

struct MainParams {
  const float* head; const float* ear; const int* fidx; const int* eidx;
  const float* cond_w1; const float* cond_b1;
  const float* cond_w2; const float* cond_b2;
  const float* f_emb; const float* d_emb;
  const float* gamma[3][8]; const float* beta[3][8]; const float* wemb[3][8];
  const float* w_comp;
  const float* to_z_w; const float* to_z_b;
  const float* dec_w1; const float* dec_b1;
  const float* dec_w2; const float* dec_b2;
  const unsigned short* Tsw;
  float* out;
};

__device__ inline float silu_gate(float g, float b, float n) {
  float t = g * n + b;
  float s = t / (1.f + __expf(-t));   // silu
  return s / (n + EPS);
}

union H8  { uint4 q; _Float16 h[8]; };
union A16 { v16h v; _Float16 h[16]; };
union B16 { v16h v; uint4 q[2]; };

// One 16x16x32 WMMA step: B from global (pre-swizzled), A = s * (held h1 regs)
__device__ __forceinline__ v8f tp_chunk(const _Float16* Tb, int c,
                                        const H8& lo, const H8& hi,
                                        _Float16 s, v8f acc) {
  B16 B;
  const uint4* bp = (const uint4*)(Tb + (size_t)c * (4 * 32 * 16));
  B.q[0] = bp[0];
  B.q[1] = bp[1];
  A16 A;
#pragma unroll
  for (int e = 0; e < 8; ++e) A.h[e] = s * lo.h[e];
#pragma unroll
  for (int e = 0; e < 8; ++e) A.h[8 + e] = s * hi.h[e];
  return __builtin_amdgcn_wmma_f32_16x16x32_f16(
      false, A.v, false, B.v, (short)0, acc, false, false);
}

__global__ __launch_bounds__(128) void so3_main_kernel(MainParams P) {
  __shared__ __align__(16) float xbuf[TILE][40];
  __shared__ __align__(16) float c1buf[TILE][24];
  __shared__ __align__(16) float cbuf[TILE][33];
  __shared__ __align__(16) _Float16 h1b[TILE][NSH];
  __shared__ __align__(16) float hnext[TILE][NSH];
  __shared__ __align__(16) float zbuf[TILE][48];
  __shared__ __align__(16) float ybuf[TILE][NSH];

  const int tid = threadIdx.x;
  const int row0 = blockIdx.x * TILE;

  // ---- stage inputs into LDS via CDNA5 async global->LDS loads ----
  for (int i = tid; i < TILE * 25; i += 128) {
    int r = i / 25, c = i % 25;
    async_load_f32_to_lds(&xbuf[r][c], &P.head[(size_t)(row0 + r) * 25 + c]);
  }
  for (int i = tid; i < TILE * 12; i += 128) {
    int r = i / 12, c = i % 12;
    async_load_f32_to_lds(&xbuf[r][25 + c], &P.ear[(size_t)(row0 + r) * 12 + c]);
  }
  wait_async0();
  __syncthreads();

  // ---- conditioning MLP -> cbuf[16][33] ----
  for (int i = tid; i < TILE * 24; i += 128) {
    int r = i / 24, o = i % 24;
    float a = P.cond_b1[o];
    for (int k = 0; k < 37; ++k) a += xbuf[r][k] * P.cond_w1[k * 24 + o];
    c1buf[r][o] = fmaxf(a, 0.f);
  }
  __syncthreads();
  for (int i = tid; i < TILE * 24; i += 128) {
    int r = i / 24, o = i % 24;
    float a = P.cond_b2[o];
    for (int k = 0; k < 24; ++k) a += c1buf[r][k] * P.cond_w2[k * 24 + o];
    cbuf[r][o] = a;
  }
  for (int i = tid; i < TILE * 6; i += 128) {
    int r = i / 6, c = i % 6;
    cbuf[r][24 + c] = P.f_emb[(size_t)P.fidx[row0 + r] * 6 + c];
  }
  for (int i = tid; i < TILE * 3; i += 128) {
    int r = i / 3, c = i % 3;
    cbuf[r][30 + c] = P.d_emb[(size_t)P.eidx[row0 + r] * 3 + c];
  }
  __syncthreads();

  const int wave = tid >> 5, lane = tid & 31;
  const int arow = lane & 15;
  const int base0 = (lane & 16) ? 16 : 0;

  // ---- 3 CG blocks ----
  for (int blk = 0; blk < 3; ++blk) {
    // norm_act + embed -> h1b (f16)
    if (blk == 0) {
      for (int i = tid; i < TILE * NSH; i += 128)
        (&h1b[0][0])[i] = (_Float16)0.f;
      __syncthreads();
      if (tid < TILE) {
        int r = tid;
        float ss = 0.f;
        for (int c = 0; c < 33; ++c) { float v = cbuf[r][c]; ss += v * v; }
        float d = sqrtf(ss / 33.f + EPS);
        float acc = 0.f;
        for (int c = 0; c < 33; ++c) {
          float x = cbuf[r][c] / d;
          float n = sqrtf(x * x + EPS);
          acc += x * silu_gate(P.gamma[0][0][c], P.beta[0][0][c], n) *
                 P.wemb[0][0][c];
        }
        h1b[r][0] = (_Float16)acc;
      }
      // l>=1 inputs are exact zeros -> h1 stays zero (matches reference).
    } else {
      int r = tid & 15, l = tid >> 4;     // 16 rows x 8 degrees = 128 threads
      int off = l * l, len = 2 * l + 1;
      float ss = 0.f;
      for (int m = 0; m < len; ++m) { float v = hnext[r][off + m]; ss += v * v; }
      float d = sqrtf(ss / (float)len + EPS);
      float s2 = 0.f;
      for (int m = 0; m < len; ++m) { float x = hnext[r][off + m] / d; s2 += x * x; }
      float n = sqrtf(s2 / (float)len + EPS);
      float gw = silu_gate(P.gamma[blk][l][0], P.beta[blk][l][0], n) *
                 P.wemb[blk][l][0];
      for (int m = 0; m < len; ++m)
        h1b[r][off + m] = (_Float16)((hnext[r][off + m] / d) * gw);
    }
    __syncthreads();

    // ---- bilinear CG tensor product as WMMA GEMM:
    //      out[16x64] = [outer(h1,h1) | h1][16 x 4160] @ Tfold[4160 x 64]
    //      Pair of chunks (2m, 2m+1) per iteration, independent accumulators,
    //      branch-free body; h1-row A slices held in registers (loop-invariant).
    const _Float16* Tb = reinterpret_cast<const _Float16*>(P.Tsw) +
                         ((size_t)blk * NCHUNK * 4 + wave) * (32 * 16) +
                         (size_t)lane * 16;
    const uint4* hr = (const uint4*)&h1b[arow][0];
    H8 r0, r1, r2, r3;                    // the 4 A-side slices, hoisted
    r0.q = hr[(0 + base0) >> 3];          // n = base0 .. base0+7
    r1.q = hr[(8 + base0) >> 3];          // n = 8+base0 ..
    r2.q = hr[(32 + base0) >> 3];         // n = 32+base0 ..
    r3.q = hr[(40 + base0) >> 3];         // n = 40+base0 ..

    v8f acc0 = {};
    v8f acc1 = {};
#pragma unroll 2
    for (int m = 0; m < 64; ++m) {
      _Float16 s = h1b[arow][m];          // broadcast scalar for this m-row
      __builtin_prefetch((const void*)(Tb + (size_t)(2 * m + 2) * (4 * 32 * 16)),
                         0, 0);
      acc0 = tp_chunk(Tb, 2 * m,     r0, r1, s, acc0);   // n in [0,32)
      acc1 = tp_chunk(Tb, 2 * m + 1, r2, r3, s, acc1);   // n in [32,64)
    }
    // linear self-path rows (j = 4096..4159): A = h1 itself
    acc0 = tp_chunk(Tb, 128, r0, r1, (_Float16)1.f, acc0);
    acc1 = tp_chunk(Tb, 129, r2, r3, (_Float16)1.f, acc1);

    __syncthreads();
#pragma unroll
    for (int r = 0; r < 8; ++r) {
      int M = r + ((lane & 16) ? 8 : 0);
      int N = (lane & 15) + wave * 16;
      hnext[M][N] = acc0[r] + acc1[r];
    }
    __syncthreads();
  }

  // ---- compressor + decoder MLP ----
  for (int i = tid; i < TILE * NSH; i += 128) {
    int r = i >> 6, k = i & 63;
    int l = 0;
    while ((l + 1) * (l + 1) <= k) ++l;
    ybuf[r][k] = hnext[r][k] * P.w_comp[l];
  }
  __syncthreads();
  for (int i = tid; i < TILE * 48; i += 128) {
    int r = i / 48, o = i % 48;
    float a = P.to_z_b[o];
    for (int k = 0; k < 64; ++k) a += ybuf[r][k] * P.to_z_w[k * 48 + o];
    zbuf[r][o] = a;
  }
  __syncthreads();
  for (int i = tid; i < TILE * NSH; i += 128) {
    int r = i >> 6, o = i & 63;
    float a = P.dec_b1[o];
    for (int k = 0; k < 48; ++k) a += zbuf[r][k] * P.dec_w1[k * 64 + o];
    ybuf[r][o] = fmaxf(a, 0.f);   // ybuf reused as y1 (vec already consumed)
  }
  __syncthreads();
  for (int i = tid; i < TILE * NSH; i += 128) {
    int r = i >> 6, o = i & 63;
    float a = P.dec_b2[o];
    for (int k = 0; k < 64; ++k) a += ybuf[r][k] * P.dec_w2[k * 64 + o];
    P.out[(size_t)(row0 + r) * 64 + o] = a;
  }
}

// ------------------------------- launcher ----------------------------------
// d_in order (JAX pytree, dict keys sorted):
//   0 head, 1 ear_feats, 2 freq_idx, 3 ear_idx,
//   params: blocks[b] -> beta[8], gamma[8], w_embed[8], w_mix[8]  (b=0..2)
//           then cond_b1, cond_b2, cond_w1, cond_w2, d_emb, dec_b1, dec_b2,
//           dec_w1, dec_w2, f_emb, to_z_b, to_z_w, w_comp
//   w3j: 154 tensors sorted by (l1,l2,l3)

extern "C" void kernel_launch(void* const* d_in, const int* in_sizes, int n_in,
                              void* d_out, int out_size, void* d_ws,
                              size_t ws_size, hipStream_t stream) {
  const int PBASE = 4;
  const int W3JBASE = 113;           // 4 + 96 block leaves + 13 top leaves
  if (n_in < W3JBASE) return;

  auto beta_p  = [&](int b, int l) { return (const float*)d_in[PBASE + b * 32 + 0 + l]; };
  auto gamma_p = [&](int b, int l) { return (const float*)d_in[PBASE + b * 32 + 8 + l]; };
  auto wemb_p  = [&](int b, int l) { return (const float*)d_in[PBASE + b * 32 + 16 + l]; };
  auto wmix_p  = [&](int b, int l) { return (const float*)d_in[PBASE + b * 32 + 24 + l]; };

  const float* cond_b1 = (const float*)d_in[100];
  const float* cond_b2 = (const float*)d_in[101];
  const float* cond_w1 = (const float*)d_in[102];
  const float* cond_w2 = (const float*)d_in[103];
  const float* d_emb   = (const float*)d_in[104];
  const float* dec_b1  = (const float*)d_in[105];
  const float* dec_b2  = (const float*)d_in[106];
  const float* dec_w1  = (const float*)d_in[107];
  const float* dec_w2  = (const float*)d_in[108];
  const float* f_emb   = (const float*)d_in[109];
  const float* to_z_b  = (const float*)d_in[110];
  const float* to_z_w  = (const float*)d_in[111];
  const float* w_comp  = (const float*)d_in[112];

  unsigned short* Tsw = (unsigned short*)d_ws;
  const int T_halves = 3 * NCHUNK * 4 * 32 * 16;   // 798720 f16 = ~1.5MB

  // 1) zero the folded tensor
  {
    int nwords = T_halves / 2;
    zero_ws_kernel<<<(nwords + 255) / 256, 256, 0, stream>>>(
        (unsigned int*)Tsw, nwords);
  }

  // 2) fold each w3j triple (with per-block w_mix) into Tsw
  {
    int pcount[8];
    for (int l = 0; l < 8; ++l) pcount[l] = 1;   // path 0 = self connection
    int t = 0;
    for (int l1 = 0; l1 <= LMAX; ++l1)
      for (int l2 = l1; l2 <= LMAX; ++l2) {
        int hi = (l1 + l2 < LMAX) ? (l1 + l2) : LMAX;
        for (int l3 = l2 - l1; l3 <= hi; ++l3) {
          int pidx = pcount[l3]++;
          int leaf = W3JBASE + t;
          ++t;
          if (leaf >= n_in) continue;
          const float* C = (const float*)d_in[leaf];
          int tot = 3 * (2 * l1 + 1) * (2 * l2 + 1) * (2 * l3 + 1);
          fold_triple_kernel<<<(tot + 255) / 256, 256, 0, stream>>>(
              Tsw, C, wmix_p(0, l3), wmix_p(1, l3), wmix_p(2, l3),
              l1, l2, l3, pidx);
        }
      }
  }

  // 3) self-connection linear rows of the folded tensor
  {
    WmPtrs W;
    for (int b = 0; b < 3; ++b)
      for (int l = 0; l < 8; ++l) W.wm[b][l] = wmix_p(b, l);
    fold_self_kernel<<<1, 3 * NSH, 0, stream>>>(Tsw, W);
  }

  // 4) main fused forward
  {
    MainParams P;
    P.head = (const float*)d_in[0];
    P.ear = (const float*)d_in[1];
    P.fidx = (const int*)d_in[2];
    P.eidx = (const int*)d_in[3];
    P.cond_w1 = cond_w1; P.cond_b1 = cond_b1;
    P.cond_w2 = cond_w2; P.cond_b2 = cond_b2;
    P.f_emb = f_emb; P.d_emb = d_emb;
    for (int b = 0; b < 3; ++b)
      for (int l = 0; l < 8; ++l) {
        P.gamma[b][l] = gamma_p(b, l);
        P.beta[b][l] = beta_p(b, l);
        P.wemb[b][l] = wemb_p(b, l);
      }
    P.w_comp = w_comp;
    P.to_z_w = to_z_w; P.to_z_b = to_z_b;
    P.dec_w1 = dec_w1; P.dec_b1 = dec_b1;
    P.dec_w2 = dec_w2; P.dec_b2 = dec_b2;
    P.Tsw = Tsw;
    P.out = (float*)d_out;

    int nrows = in_sizes[2];               // B (freq_idx element count)
    int nblocks = nrows / TILE;
    so3_main_kernel<<<nblocks, 128, 0, stream>>>(P);
  }
}
